// LSHAttention_13280038879438
// MI455X (gfx1250) — matrix operand order repaired and checked by reading, not compiled, not run on hardware
//
#include <hip/hip_runtime.h>
#include <hip/hip_bf16.h>
#include <math.h>

#define NB_B    16
#define NB_L    4096
#define NB_DK   64
#define NB_R    4
#define NB_NB   64      // n_buckets (also bucket value range)
#define NB_BL   64      // bucket_len
#define NEG_BIG  (-1e9f)
#define NEG_SELF (-1e5f)

typedef __attribute__((ext_vector_type(16))) __bf16 v16bf;
typedef __attribute__((ext_vector_type(8)))  float  v8f;
typedef __attribute__((ext_vector_type(8)))  unsigned short ushort8;

struct U16x16 { ushort8 lo; ushort8 hi; };

__device__ __forceinline__ unsigned short f2bf(float f) {
    unsigned int u = __builtin_bit_cast(unsigned int, f);
    u += 0x7fffu + ((u >> 16) & 1u);   // round-to-nearest-even
    return (unsigned short)(u >> 16);
}
__device__ __forceinline__ unsigned int pack2bf(float a, float b) {
    return (unsigned int)f2bf(a) | ((unsigned int)f2bf(b) << 16);
}
__device__ __forceinline__ v16bf ld_frag(const unsigned short* p0, const unsigned short* p1) {
    U16x16 t;
    t.lo = *(const ushort8*)p0;
    t.hi = *(const ushort8*)p1;
    return __builtin_bit_cast(v16bf, t);
}

// ---------------------------------------------------------------- K1: normalize rand matrix columns over d
__global__ void k_rmnorm(const float* __restrict__ rm, float* __restrict__ rmn) {
    __shared__ float red[64];
    int c = blockIdx.x;             // B*R*32 columns
    int n = c & 31;
    int r = (c >> 5) & 3;
    int b = c >> 7;
    int d = threadIdx.x;            // 64 threads
    int idx = b * (64 * NB_R * 32) + d * (NB_R * 32) + r * 32 + n;
    float v = rm[idx];
    red[d] = v * v;
    __syncthreads();
    for (int s = 32; s > 0; s >>= 1) {
        if (d < s) red[d] += red[d + s];
        __syncthreads();
    }
    float inv = 1.0f / sqrtf(red[0]);
    rmn[idx] = v * inv;
}

// ---------------------------------------------------------------- K2: LSH hash (argmax over [mm,-mm])
__global__ void k_hash(const float* __restrict__ query, const float* __restrict__ rmn,
                       int* __restrict__ hbuck) {
    __shared__ float qrow[64];
    int bx = blockIdx.x;            // B*L
    int l = bx & (NB_L - 1);
    int b = bx / NB_L;
    int t = threadIdx.x;            // 128 threads
    if (t < 64) qrow[t] = query[((size_t)(b * NB_L + l)) * 64 + t];
    __syncthreads();
    int r = t >> 5;
    int nl = t & 31;
    const float* rb = rmn + (size_t)b * 8192 + r * 32 + nl;
    float acc = 0.0f;
#pragma unroll 8
    for (int d = 0; d < 64; ++d) acc += qrow[d] * rb[d * 128];
    float bv = acc; int bi = nl;
    if (-acc > bv) { bv = -acc; bi = nl + 32; }      // strict > keeps lower index on ties
    for (int m = 16; m > 0; m >>= 1) {
        float ov = __shfl_xor(bv, m, 32);
        int   oi = __shfl_xor(bi, m, 32);
        if (ov > bv || (ov == bv && oi < bi)) { bv = ov; bi = oi; }
    }
    if (nl == 0) hbuck[(size_t)(b * NB_R + r) * NB_L + l] = bi;
}

// ---------------------------------------------------------------- K3: stable counting sort per (b,r)
__global__ void k_sort(const int* __restrict__ hbuck,
                       int* __restrict__ hidx, int* __restrict__ bsrt, int* __restrict__ chof) {
    __shared__ int tileLDS[512];
    __shared__ int cnts[64];
    int bid = blockIdx.x;           // B*R
    size_t base = (size_t)bid * NB_L;
    int v = threadIdx.x;            // 64 threads, thread v owns bucket value v
    int cnt = 0;
    for (int tgt = 0; tgt < NB_L; tgt += 512) {
        for (int i = v; i < 512; i += 64) tileLDS[i] = hbuck[base + tgt + i];
        __syncthreads();
        for (int i = 0; i < 512; ++i) cnt += (tileLDS[i] == v);
        __syncthreads();
    }
    cnts[v] = cnt;
    __syncthreads();
    if (v == 0) {
        int run = 0;
        for (int k = 0; k < 64; ++k) { int c = cnts[k]; cnts[k] = run; run += c; }
    }
    __syncthreads();
    int pos = cnts[v];
    for (int tgt = 0; tgt < NB_L; tgt += 512) {
        for (int i = v; i < 512; i += 64) tileLDS[i] = hbuck[base + tgt + i];
        __syncthreads();
        for (int i = 0; i < 512; ++i) {
            if (tileLDS[i] == v) {
                int l = tgt + i;
                hidx[base + pos] = l;
                bsrt[base + pos] = v;
                chof[base + l]   = pos >> 6;
                ++pos;
            }
        }
        __syncthreads();
    }
}

// ---------------------------------------------------------------- K4: WMMA attention core (one (b,chunk,round) tile)
__global__ void __launch_bounds__(256) k_attn(
        const float* __restrict__ query, const float* __restrict__ value,
        const int* __restrict__ hidx, const int* __restrict__ bsrt, const int* __restrict__ chof,
        float* __restrict__ attn_buf, float* __restrict__ lse_buf) {
    __shared__ __align__(16) unsigned char smem[62976];
    unsigned short* qs  = (unsigned short*)smem;            // [64][64] bf16, GEMM1 A
    unsigned short* ks  = (unsigned short*)(smem + 8192);   // [128][64] bf16 normalized keys
    unsigned short* vt  = (unsigned short*)smem;            // [64][136] bf16 V^T  (aliases qs/ks after GEMM1)
    float*          sco = (float*)(smem + 24576);           // [64][132] f32 scores
    unsigned short* smx = (unsigned short*)(smem + 24576);  // [64][128] bf16 softmax (aliases sco, staged)
    int* meta  = (int*)(smem + 58368);
    int* qi_s  = meta;          // 64
    int* qb_s  = meta + 64;     // 64
    int* qch_s = meta + 128;    // 64*4
    int* ki_s  = meta + 384;    // 128
    int* kb_s  = meta + 512;    // 128
    int* kch_s = meta + 640;    // 128*4

    int bx = blockIdx.x;        // B*NB*R
    int r  = bx & 3;
    int n  = (bx >> 2) & 63;
    int b  = bx >> 8;
    int tid = threadIdx.x;
    int w   = tid >> 5;         // wave 0..7
    int ln  = tid & 31;
    size_t hbase = (size_t)(b * NB_R + r) * NB_L;

    // --- load Q tile (fold 1/sqrt(dk)=0.125 into bf16) -------------------
    for (int it = 0; it < 8; ++it) {
        int i = w + 8 * it;
        int p = n * 64 + i;
        int l = hidx[hbase + p];
        const float2 qv = ((const float2*)(query + (size_t)(b * NB_L + l) * 64))[ln];
        *(unsigned int*)&qs[i * 64 + 2 * ln] = pack2bf(qv.x * 0.125f, qv.y * 0.125f);
        if (ln == 0) { qi_s[i] = l; qb_s[i] = bsrt[hbase + p]; }
        if (ln < 4)  qch_s[i * 4 + ln] = chof[(size_t)(b * NB_R + ln) * NB_L + l];
    }
    // --- load K tile (gathered Q rows, L2-normalized) --------------------
    for (int it = 0; it < 16; ++it) {
        int j  = w + 8 * it;
        int sp = (j < 64) ? (((n + NB_NB - 1) & 63) * 64 + j) : (n * 64 + (j - 64));
        int kl = hidx[hbase + sp];
        const float2 kv = ((const float2*)(query + (size_t)(b * NB_L + kl) * 64))[ln];
        float ss = kv.x * kv.x + kv.y * kv.y;
        for (int m = 16; m > 0; m >>= 1) ss += __shfl_xor(ss, m, 32);
        float inv = 1.0f / fmaxf(sqrtf(ss), 1e-12f);
        *(unsigned int*)&ks[j * 64 + 2 * ln] = pack2bf(kv.x * inv, kv.y * inv);
        if (ln == 0) { ki_s[j] = kl; kb_s[j] = bsrt[hbase + sp]; }
        if (ln < 4)  kch_s[j * 4 + ln] = chof[(size_t)(b * NB_R + ln) * NB_L + kl];
    }
    __syncthreads();

    // --- GEMM1: scores[64][128] = Qs @ Ks^T  (wmma bf16, K=64) -----------
    int mpr  = ln & 15;
    int half = ln >> 4;
    for (int t2 = 0; t2 < 4; ++t2) {
        int tile = w * 4 + t2;
        int mt = tile >> 3, nt = tile & 7;
        v8f acc = {};
#pragma unroll
        for (int kk = 0; kk < 2; ++kk) {
            int k0 = kk * 32;
            const unsigned short* ar = qs + (mt * 16 + mpr) * 64 + k0 + half * 8;
            const unsigned short* br = ks + (nt * 16 + mpr) * 64 + k0 + half * 8;
            v16bf a = ld_frag(ar, ar + 16);
            v16bf bm = ld_frag(br, br + 16);
            acc = __builtin_amdgcn_wmma_f32_16x16x32_bf16(false, a, false, bm,
                                                          (short)0, acc, false, false);
        }
#pragma unroll
        for (int vv = 0; vv < 8; ++vv)
            sco[(mt * 16 + vv + 8 * half) * 132 + nt * 16 + mpr] = acc[vv];
    }
    __syncthreads();

    // --- load V^T into aliased region (bf16, stride 136) -----------------
    for (int it = 0; it < 16; ++it) {
        int j  = w + 8 * it;
        int kl = ki_s[j];
        const float2 vv2 = ((const float2*)(value + (size_t)(b * NB_L + kl) * 64))[ln];
        vt[(2 * ln) * 136 + j]     = f2bf(vv2.x);
        vt[(2 * ln + 1) * 136 + j] = f2bf(vv2.y);
    }

    // --- masked, dup-count-adjusted softmax ------------------------------
    float buf[4][4];
    for (int rr = 0; rr < 8; ++rr) {
        if (rr == 4) {
            __syncthreads();                       // all score reads (rows 0..31) done
            for (int q = 0; q < 4; ++q) {
                int row = w + 8 * q;
                for (int c = 0; c < 4; ++c)
                    smx[row * 128 + ln + 32 * c] = f2bf(buf[q][c]);
            }
        }
        int i = w + 8 * rr;
        int qiv = qi_s[i], qbv = qb_s[i];
        int qc0 = qch_s[i * 4 + 0], qc1 = qch_s[i * 4 + 1],
            qc2 = qch_s[i * 4 + 2], qc3 = qch_s[i * 4 + 3];
        float sv[4]; int cntv[4];
        float M = -3.4e38f;
        for (int c = 0; c < 4; ++c) {
            int j = ln + 32 * c;
            float s = sco[i * 132 + j];
            int kjv = ki_s[j];
            if (kb_s[j] != qbv) s = NEG_BIG;
            if (qiv < kjv)      s = NEG_BIG;
            if (qiv == kjv)     s = NEG_SELF;
            int kc0 = kch_s[j * 4 + 0], kc1 = kch_s[j * 4 + 1],
                kc2 = kch_s[j * 4 + 2], kc3 = kch_s[j * 4 + 3];
            int cnt = (kc0 == qc0 || kc0 == ((qc0 + 63) & 63))
                    + (kc1 == qc1 || kc1 == ((qc1 + 63) & 63))
                    + (kc2 == qc2 || kc2 == ((qc2 + 63) & 63))
                    + (kc3 == qc3 || kc3 == ((qc3 + 63) & 63));
            sv[c] = s; cntv[c] = cnt;
            M = fmaxf(M, s);
        }
        for (int m = 16; m > 0; m >>= 1) M = fmaxf(M, __shfl_xor(M, m, 32));
        float S = 0.0f;
        for (int c = 0; c < 4; ++c) S += __expf(sv[c] - M);
        for (int m = 16; m > 0; m >>= 1) S += __shfl_xor(S, m, 32);
        float lse = M + logf(S);
        if (ln == 0) lse_buf[hbase + qiv] = lse;
        for (int c = 0; c < 4; ++c) {
            float smv = __expf(sv[c] - M) / (S * (float)cntv[c]);
            if (rr < 4) buf[rr][c] = smv;
            else        smx[i * 128 + ln + 32 * c] = f2bf(smv);
        }
    }
    __syncthreads();

    // --- GEMM2: attn[64][64] = SM @ V  (wmma bf16, K=128) ----------------
    for (int t2 = 0; t2 < 2; ++t2) {
        int tile = w * 2 + t2;
        int mt = tile >> 2, nt = tile & 3;
        v8f acc = {};
#pragma unroll
        for (int kk = 0; kk < 4; ++kk) {
            int k0 = kk * 32;
            const unsigned short* ar = smx + (mt * 16 + mpr) * 128 + k0 + half * 8;
            const unsigned short* br = vt  + (nt * 16 + mpr) * 136 + k0 + half * 8;
            v16bf a = ld_frag(ar, ar + 16);
            v16bf bm = ld_frag(br, br + 16);
            acc = __builtin_amdgcn_wmma_f32_16x16x32_bf16(false, a, false, bm,
                                                          (short)0, acc, false, false);
        }
#pragma unroll
        for (int vv = 0; vv < 8; ++vv) {
            int row = mt * 16 + vv + 8 * half;
            attn_buf[(hbase + (size_t)qi_s[row]) * 64 + nt * 16 + mpr] = acc[vv];
        }
    }
}

// ---------------------------------------------------------------- K5: per-(b,r) softmax-over-L stats of lse
__global__ void k_lsered(const float* __restrict__ lse_buf,
                         float* __restrict__ mx, float* __restrict__ sums) {
    __shared__ float red[256];
    int bid = blockIdx.x;
    size_t base = (size_t)bid * NB_L;
    int t = threadIdx.x;
    float m = -3.4e38f;
    for (int i = t; i < NB_L; i += 256) m = fmaxf(m, lse_buf[base + i]);
    red[t] = m; __syncthreads();
    for (int s = 128; s > 0; s >>= 1) { if (t < s) red[t] = fmaxf(red[t], red[t + s]); __syncthreads(); }
    float M = red[0]; __syncthreads();
    float acc = 0.0f;
    for (int i = t; i < NB_L; i += 256) acc += __expf(lse_buf[base + i] - M);
    red[t] = acc; __syncthreads();
    for (int s = 128; s > 0; s >>= 1) { if (t < s) red[t] += red[t + s]; __syncthreads(); }
    if (t == 0) { mx[bid] = M; sums[bid] = red[0]; }
}

// ---------------------------------------------------------------- K6: combine rounds with weights
__global__ void k_combine(const float* __restrict__ attn_buf, const float* __restrict__ lse_buf,
                          const float* __restrict__ mx, const float* __restrict__ sums,
                          float* __restrict__ out) {
    int bx = blockIdx.x;            // B*L
    int l = bx & (NB_L - 1);
    int b = bx >> 12;
    int d = threadIdx.x;            // 64
    float acc = 0.0f;
    for (int r = 0; r < NB_R; ++r) {
        int br = b * NB_R + r;
        float wr = __expf(lse_buf[(size_t)br * NB_L + l] - mx[br]) / sums[br];
        acc += attn_buf[((size_t)br * NB_L + l) * 64 + d] * wr;
    }
    out[((size_t)(b * NB_L + l)) * 64 + d] = acc;
}

// ---------------------------------------------------------------- launch
extern "C" void kernel_launch(void* const* d_in, const int* in_sizes, int n_in,
                              void* d_out, int out_size, void* d_ws, size_t ws_size,
                              hipStream_t stream) {
    (void)in_sizes; (void)n_in; (void)out_size; (void)ws_size;
    const float* query = (const float*)d_in[0];
    const float* value = (const float*)d_in[1];
    const float* rm    = (const float*)d_in[2];
    float* out = (float*)d_out;

    char* ws = (char*)d_ws;
    float* rmn      = (float*)(ws + 0);            //   512 KB  [B,64,R,32]
    int*   hbuck    = (int*)  (ws + 524288);       //  1024 KB  [B,R,L]
    int*   hidx     = (int*)  (ws + 1572864);      //  1024 KB
    int*   bsrt     = (int*)  (ws + 2621440);      //  1024 KB
    int*   chof     = (int*)  (ws + 3670016);      //  1024 KB
    float* lse_buf  = (float*)(ws + 4718592);      //  1024 KB  [B,R,L]
    float* mx       = (float*)(ws + 5767168);      //   1 KB pad
    float* sums     = (float*)(ws + 5768192);      //   1 KB pad
    float* attn_buf = (float*)(ws + 5769216);      //    64 MB  [B,R,L,64]

    k_rmnorm <<<NB_B * NB_R * 32, 64, 0, stream>>>(rm, rmn);
    k_hash   <<<NB_B * NB_L, 128, 0, stream>>>(query, rmn, hbuck);
    k_sort   <<<NB_B * NB_R, 64, 0, stream>>>(hbuck, hidx, bsrt, chof);
    k_attn   <<<NB_B * NB_NB * NB_R, 256, 0, stream>>>(query, value, hidx, bsrt, chof,
                                                       attn_buf, lse_buf);
    k_lsered <<<NB_B * NB_R, 256, 0, stream>>>(lse_buf, mx, sums);
    k_combine<<<NB_B * NB_L, 64, 0, stream>>>(attn_buf, lse_buf, mx, sums, out);
}
